// TransformerBlock_8349416423522
// MI455X (gfx1250) — matrix operand order repaired and checked by reading, not compiled, not run on hardware
//
#include <hip/hip_runtime.h>
#include <stdint.h>

// ---------------------------------------------------------------------------
// Types for CDNA5 WMMA / TDM (gfx1250, wave32)
// ---------------------------------------------------------------------------
typedef __attribute__((ext_vector_type(8)))  __bf16 v8bf;
typedef __attribute__((ext_vector_type(16))) __bf16 v16bf;
typedef __attribute__((ext_vector_type(8)))  float  v8f;
typedef __attribute__((ext_vector_type(4)))  unsigned int u32x4;
typedef __attribute__((ext_vector_type(4)))  int i32x4;
typedef __attribute__((ext_vector_type(8)))  int i32x8;

#define WAVE 32

// D = A(16x32 bf16) x B(32x16 bf16) + C(16x16 f32)
__device__ __forceinline__ v8f wmma_bf16(v16bf a, v16bf b, v8f c) {
  return __builtin_amdgcn_wmma_f32_16x16x32_bf16(
      /*neg_a=*/false, a, /*neg_b=*/false, b,
      /*c_mod=*/(short)0, c, /*reuse_a=*/false, /*reuse_b=*/false);
}

// B-fragment (and P/V fragments): lane holds 16 contiguous bf16 at p.
__device__ __forceinline__ v16bf ld16(const __bf16* p) {
  v16bf f;
  ((v8bf*)&f)[0] = *(const v8bf*)(p);
  ((v8bf*)&f)[1] = *(const v8bf*)(p + 8);
  return f;
}
// A-fragment: lane holds row ptr prow; elements kb..kb+7 and 16+kb..16+kb+7
// (kb = 8*(lane>>4)) per the 16-bit A 16x32 VGPR layout.
__device__ __forceinline__ v16bf ldA(const __bf16* prow, int kb) {
  v16bf f;
  ((v8bf*)&f)[0] = *(const v8bf*)(prow + kb);
  ((v8bf*)&f)[1] = *(const v8bf*)(prow + 16 + kb);
  return f;
}

// ---------------------------------------------------------------------------
// Tensor Data Mover: DMA one [128 rows x 32 cols] bf16 tile (row-major,
// row stride K elements) from global into LDS with a 16B pad after each 64B
// row (-> LDS row stride = 40 bf16, matching the fragment readers).
// D# layout per CDNA5 ISA sec 8.3/8.4. Issued by one wave; EXEC ignored.
// 6-arg builtin: (u32x4 g0, i32x8 g1, i32x4 g2, i32x4 g3, i32x8 g4, i32 cpol)
// ---------------------------------------------------------------------------
__device__ __forceinline__ void tdm_load_tile_128x32(
    unsigned long long gaddr, unsigned lds_off, int rows, int K) {
  u32x4 g0;
  g0.x = 1u;                                    // count=1 (valid user D#)
  g0.y = lds_off;                               // LDS byte address
  g0.z = (unsigned)(gaddr & 0xFFFFFFFFull);     // global_addr[31:0]
  g0.w = (unsigned)((gaddr >> 32) & 0x01FFFFFFull) | (2u << 30);  // type=2
  i32x8 g1;
  // data_size=1 (2B), pad_enable=1, pad_interval=3 (16 DW = 64B),
  // pad_amount=3 (4 DW = 16B); workgroup_mask=0 (no cluster)
  g1[0] = (1 << 16) | (1 << 20) | (3 << 22) | (3 << 25);
  g1[1] = (K & 0xFFFF) << 16;                             // tensor_dim0[15:0]
  g1[2] = (int)(((unsigned)K >> 16) | ((unsigned)(rows & 0xFFFF) << 16));
  g1[3] = (int)(((unsigned)rows >> 16) | (32u << 16));    // dim1 hi | tile_dim0=32
  g1[4] = 128;                                            // tile_dim1=128, tile_dim2=0
  g1[5] = K;                                              // tensor_dim0_stride[31:0]
  g1[6] = 0;                                              // stride0 hi | stride1 lo
  g1[7] = 0;
  i32x4 z4 = {0, 0, 0, 0};  // 2D tensor: groups 2/3 unused
  i32x8 z8 = {0, 0, 0, 0, 0, 0, 0, 0};
  __builtin_amdgcn_tensor_load_to_lds(g0, g1, z4, z4, z8, 0);
}

// ---------------------------------------------------------------------------
// Elementwise helpers
// ---------------------------------------------------------------------------
__global__ void cvt_f32_bf16_kernel(const float* __restrict__ in,
                                    __bf16* __restrict__ out, int n) {
  int i = blockIdx.x * blockDim.x + threadIdx.x;
  if (i < n) out[i] = (__bf16)in[i];
}

// RMSNorm: one block per row of [rows, 768]; fp32 in -> bf16 out
__global__ __launch_bounds__(256) void rmsnorm_kernel(
    const float* __restrict__ x, const float* __restrict__ w,
    __bf16* __restrict__ out, int D) {
  int row = blockIdx.x;
  const float* xr = x + (size_t)row * D;
  __shared__ float red[256];
  float ss = 0.f;
  for (int i = threadIdx.x; i < D; i += 256) {
    float v = xr[i];
    ss += v * v;
  }
  red[threadIdx.x] = ss;
  __syncthreads();
  for (int s = 128; s > 0; s >>= 1) {
    if (threadIdx.x < (unsigned)s) red[threadIdx.x] += red[threadIdx.x + s];
    __syncthreads();
  }
  float rinv = rsqrtf(red[0] / (float)D + 1e-5f);
  for (int i = threadIdx.x; i < D; i += 256)
    out[(size_t)row * D + i] = (__bf16)(xr[i] * rinv * w[i]);
}

// RoPE on [B,S,H,DH] fp32 -> bf16 ; pairs (2k, 2k+1), ang = s/theta^(2k/DH)
// outScale lets us fold the attention 1/sqrt(DH) into Q (scores = (Q/8)K^T).
__global__ void rope_kernel(const float* __restrict__ X,
                            __bf16* __restrict__ O, int total, float outScale) {
  int i = blockIdx.x * blockDim.x + threadIdx.x;
  if (i >= total) return;
  int kk = i & 31;  // DH/2 = 32
  int t = i >> 5;
  int h = t % 12;
  t /= 12;
  int s = t & 2047;
  int b = t >> 11;
  size_t base = ((size_t)(b * 2048 + s)) * 768 + h * 64 + 2 * kk;
  float fr = __powf(10000.0f, -(float)(2 * kk) * (1.0f / 64.0f));
  float sn, cs;
  __sincosf((float)s * fr, &sn, &cs);
  float x0 = X[base], x1 = X[base + 1];
  O[base]     = (__bf16)((cs * x0 - sn * x1) * outScale);
  O[base + 1] = (__bf16)((sn * x0 + cs * x1) * outScale);
}

// V [B,S,H,DH] fp32 -> VT [B,H,DH,S] bf16
__global__ void vtrans_kernel(const float* __restrict__ V,
                              __bf16* __restrict__ VT, int total) {
  int i = blockIdx.x * blockDim.x + threadIdx.x;
  if (i >= total) return;
  int col = i % 768;
  int row = i / 768;
  int h = col >> 6, e = col & 63;
  int b = row >> 11, s = row & 2047;
  VT[(((size_t)(b * 12 + h)) * 64 + e) * 2048 + s] = (__bf16)V[i];
}

// g = silu(u1) * u3 -> bf16
__global__ void silu_mul_kernel(const float* __restrict__ U1,
                                const float* __restrict__ U3,
                                __bf16* __restrict__ G, int n) {
  int i = blockIdx.x * blockDim.x + threadIdx.x;
  if (i >= n) return;
  float u = U1[i];
  float sg = 1.0f / (1.0f + __expf(-u));
  G[i] = (__bf16)(u * sg * U3[i]);
}

// ---------------------------------------------------------------------------
// Tiled WMMA GEMM: C[M,N] (f32, optional +resid) = A[M,K] (bf16) * W[N,K]^T
// Block tile 128x128, BK=32, 8 waves (4x2), each wave 32x64 (2x4 frags).
// Tile staging via Tensor Data Mover: wave0 DMAs the A tile, wave1 the W
// tile, each waits TENSORcnt, workgroup barrier releases compute.
// ---------------------------------------------------------------------------
template <bool RESID>
__global__ __launch_bounds__(256) void gemm_bf16_kernel(
    const __bf16* __restrict__ A, const __bf16* __restrict__ W,
    float* __restrict__ C, const float* __restrict__ resid,
    int M, int N, int K) {
  __shared__ __bf16 As[128 * 40];  // 40 bf16 stride = 64B row + 16B TDM pad
  __shared__ __bf16 Ws[128 * 40];
  int tid = threadIdx.x;
  int wid = tid >> 5, lane = tid & 31;
  int g = lane >> 4, hl = lane & 15;
  int mw = wid >> 1, nw = wid & 1;
  int m0 = blockIdx.y * 128, n0 = blockIdx.x * 128;
  int waveM = mw * 32, waveN = nw * 64;

  unsigned long long aBase =
      (unsigned long long)(uintptr_t)A + (unsigned long long)((size_t)m0 * K) * 2ull;
  unsigned long long wBase =
      (unsigned long long)(uintptr_t)W + (unsigned long long)((size_t)n0 * K) * 2ull;
  unsigned ldsA = (unsigned)(uintptr_t)(&As[0]);  // low 32 bits = LDS offset
  unsigned ldsW = (unsigned)(uintptr_t)(&Ws[0]);

  v8f acc[2][4];
#pragma unroll
  for (int t = 0; t < 2; ++t)
#pragma unroll
    for (int u = 0; u < 4; ++u)
#pragma unroll
      for (int r = 0; r < 8; ++r) acc[t][u][r] = 0.0f;

  for (int kt = 0; kt < K; kt += 32) {
    if (wid == 0) {
      tdm_load_tile_128x32(aBase + (unsigned long long)kt * 2ull, ldsA, M, K);
      __builtin_amdgcn_s_wait_tensorcnt(0);
    } else if (wid == 1) {
      tdm_load_tile_128x32(wBase + (unsigned long long)kt * 2ull, ldsW, N, K);
      __builtin_amdgcn_s_wait_tensorcnt(0);
    }
    __syncthreads();

    v16bf af[2], bfr[4];
#pragma unroll
    for (int t = 0; t < 2; ++t)
      af[t] = ldA(&As[(waveM + t * 16 + hl) * 40], g * 8);
#pragma unroll
    for (int u = 0; u < 4; ++u)
      bfr[u] = ldA(&Ws[(waveN + u * 16 + hl) * 40], g * 8);
#pragma unroll
    for (int t = 0; t < 2; ++t)
#pragma unroll
      for (int u = 0; u < 4; ++u)
        acc[t][u] = wmma_bf16(af[t], bfr[u], acc[t][u]);
    __syncthreads();
  }

#pragma unroll
  for (int t = 0; t < 2; ++t) {
    int mbase = m0 + waveM + t * 16 + 8 * g;  // C layout: VGPR r -> M = r+8g
#pragma unroll
    for (int u = 0; u < 4; ++u) {
      int ncol = n0 + waveN + u * 16 + hl;    // lane -> N = lane&15
      size_t base = (size_t)mbase * N + ncol;
      float* cp = C + base;
      if constexpr (RESID) {
        const float* rp = resid + base;
#pragma unroll
        for (int r = 0; r < 8; ++r) {
          *cp = acc[t][u][r] + *rp;
          cp += N;
          rp += N;
        }
      } else {
#pragma unroll
        for (int r = 0; r < 8; ++r) {
          *cp = acc[t][u][r];
          cp += N;
        }
      }
    }
  }
}

// ---------------------------------------------------------------------------
// Flash attention (causal), transposed-score formulation.
// Q (pre-scaled by 1/8), K bf16 [B,S,H*DH]; VT bf16 [B,H,DH,S]; O bf16.
// Block = 8 waves; each wave owns 16 query rows. Per 32-key tile:
//   St = K_tile(32x64) @ Q^T  (C-layout: lane = q column, rows = keys)
//   -> P fragment for P@V is exactly St's per-lane data (no relayout).
// Only the final tile of each wave straddles the causal diagonal, so the
// mask compares are emitted only in the MASK=true instantiation.
// ---------------------------------------------------------------------------
template <bool MASK>
__device__ __forceinline__ void flash_tile(
    const __bf16* Kb, const __bf16* Vb, const v16bf (&qf)[2], v8f (&acc)[4],
    float& m_i, float& l_i, int kbase, int qglob, int g, int hl) {
  constexpr int S = 2048, D = 768;
  v8f st[2];
#pragma unroll
  for (int t = 0; t < 2; ++t) {
#pragma unroll
    for (int r = 0; r < 8; ++r) st[t][r] = 0.0f;
    const __bf16* krow = Kb + (size_t)(kbase + t * 16 + hl) * D;
    v16bf ka0 = ldA(krow, g * 8);       // e 0..31
    v16bf ka1 = ldA(krow + 32, g * 8);  // e 32..63
    st[t] = wmma_bf16(ka0, qf[0], st[t]);
    st[t] = wmma_bf16(ka1, qf[1], st[t]);
  }
  // causal mask (last tile only) + local max over this lane's keys
  float m_loc = -1e30f;
#pragma unroll
  for (int t = 0; t < 2; ++t)
#pragma unroll
    for (int r = 0; r < 8; ++r) {
      float sv = st[t][r];
      if constexpr (MASK) {
        int kg = kbase + t * 16 + r + 8 * g;
        sv = (kg <= qglob) ? sv : -1e30f;
        st[t][r] = sv;
      }
      m_loc = fmaxf(m_loc, sv);
    }
  m_loc = fmaxf(m_loc, __shfl_xor(m_loc, 16, WAVE));  // other key half
  float m_new = fmaxf(m_i, m_loc);
  float l_loc = 0.0f;
  v16bf pa;  // P fragment for P@V: f[0..7]=st0 keys, f[8..15]=st1 keys
#pragma unroll
  for (int r = 0; r < 8; ++r) {
    float p0 = __expf(st[0][r] - m_new);
    float p1 = __expf(st[1][r] - m_new);
    l_loc += p0 + p1;
    pa[r]     = (__bf16)p0;
    pa[8 + r] = (__bf16)p1;
  }
  l_loc += __shfl_xor(l_loc, 16, WAVE);
  float corr = __expf(m_i - m_new);
  l_i = l_i * corr + l_loc;
  m_i = m_new;
  // rescale accumulators: acc element r belongs to query row r+8g
#pragma unroll
  for (int r = 0; r < 8; ++r) {
    float cr = __shfl(corr, r + 8 * g, WAVE);
#pragma unroll
    for (int u = 0; u < 4; ++u) acc[u][r] *= cr;
  }
  // ctx += P(16q x 32k) @ V(32k x 64e), 4 n-tiles of 16
#pragma unroll
  for (int u = 0; u < 4; ++u) {
    v16bf vb = ldA(Vb + (size_t)(u * 16 + hl) * S + kbase, g * 8);
    acc[u] = wmma_bf16(pa, vb, acc[u]);
  }
}

__global__ __launch_bounds__(256) void flash_attn_kernel(
    const __bf16* __restrict__ Q, const __bf16* __restrict__ Km,
    const __bf16* __restrict__ VT, __bf16* __restrict__ O) {
  constexpr int S = 2048, D = 768, H = 12, DH = 64;
  int bh = blockIdx.y;
  int b = bh / H, h = bh % H;
  int wid = threadIdx.x >> 5;
  int lane = threadIdx.x & 31;
  int g = lane >> 4, hl = lane & 15;
  int q0 = blockIdx.x * 128 + wid * 16;

  const __bf16* Qb = Q + ((size_t)b * S) * D + h * DH;
  const __bf16* Kb = Km + ((size_t)b * S) * D + h * DH;
  const __bf16* Vb = VT + ((size_t)bh * DH) * S;

  // Q as B-fragments (lane = q row), e split into 2 chunks of 32
  v16bf qf[2];
#pragma unroll
  for (int j = 0; j < 2; ++j)
    qf[j] = ldA(Qb + (size_t)(q0 + hl) * D + j * 32, g * 8);

  v8f acc[4];
#pragma unroll
  for (int u = 0; u < 4; ++u)
#pragma unroll
    for (int r = 0; r < 8; ++r) acc[u][r] = 0.0f;

  float m_i = -1e30f, l_i = 0.0f;
  int qglob = q0 + hl;
  int tiles = (q0 + 47) >> 5;      // ceil((q0+16)/32); >= 1
  int fullEnd = (tiles - 1) * 32;  // kbase of the single diagonal tile

  for (int kbase = 0; kbase < fullEnd; kbase += 32)
    flash_tile<false>(Kb, Vb, qf, acc, m_i, l_i, kbase, qglob, g, hl);
  flash_tile<true>(Kb, Vb, qf, acc, m_i, l_i, fullEnd, qglob, g, hl);

  float linv = 1.0f / l_i;
#pragma unroll
  for (int r = 0; r < 8; ++r) {
    float lv = __shfl(linv, r + 8 * g, WAVE);
    int qr = q0 + r + 8 * g;
#pragma unroll
    for (int u = 0; u < 4; ++u)
      O[((size_t)b * S + qr) * D + h * DH + u * 16 + hl] =
          (__bf16)(acc[u][r] * lv);
  }
}

// ---------------------------------------------------------------------------
// Launch
// ---------------------------------------------------------------------------
extern "C" void kernel_launch(void* const* d_in, const int* in_sizes, int n_in,
                              void* d_out, int out_size, void* d_ws,
                              size_t ws_size, hipStream_t stream) {
  (void)in_sizes; (void)n_in; (void)out_size; (void)ws_size;
  constexpr int B = 4, S = 2048, D = 768, H = 12, DF = 2048;
  constexpr int M = B * S;  // 8192

  const float* x   = (const float*)d_in[0];
  const float* ln1 = (const float*)d_in[1];
  const float* ln2 = (const float*)d_in[2];
  const float* pq  = (const float*)d_in[3];
  const float* pk  = (const float*)d_in[4];
  const float* pv  = (const float*)d_in[5];
  const float* po  = (const float*)d_in[6];
  const float* w1  = (const float*)d_in[7];
  const float* w2  = (const float*)d_in[8];
  const float* w3  = (const float*)d_in[9];
  float* out = (float*)d_out;

  char* ws = (char*)d_ws;
  size_t off = 0;
  auto alloc = [&](size_t bytes) -> void* {
    void* p = ws + off;
    off = (off + bytes + 255) & ~(size_t)255;
    return p;
  };

  __bf16* wqB = (__bf16*)alloc((size_t)D * D * 2);
  __bf16* wkB = (__bf16*)alloc((size_t)D * D * 2);
  __bf16* wvB = (__bf16*)alloc((size_t)D * D * 2);
  __bf16* woB = (__bf16*)alloc((size_t)D * D * 2);
  __bf16* w1B = (__bf16*)alloc((size_t)DF * D * 2);
  __bf16* w3B = (__bf16*)alloc((size_t)DF * D * 2);
  __bf16* w2B = (__bf16*)alloc((size_t)D * DF * 2);
  __bf16* h1  = (__bf16*)alloc((size_t)M * D * 2);
  float*  qf  = (float*)alloc((size_t)M * D * 4);
  float*  kf  = (float*)alloc((size_t)M * D * 4);
  float*  vf  = (float*)alloc((size_t)M * D * 4);
  __bf16* qb  = (__bf16*)alloc((size_t)M * D * 2);
  __bf16* kb  = (__bf16*)alloc((size_t)M * D * 2);
  __bf16* vT  = (__bf16*)alloc((size_t)M * D * 2);
  __bf16* ctx = (__bf16*)alloc((size_t)M * D * 2);
  float*  xmid = (float*)alloc((size_t)M * D * 4);
  __bf16* h2  = (__bf16*)alloc((size_t)M * D * 2);
  float*  u1  = (float*)alloc((size_t)M * DF * 4);
  float*  u3  = (float*)alloc((size_t)M * DF * 4);
  __bf16* gb  = (__bf16*)alloc((size_t)M * DF * 2);

  auto cdiv = [](int a, int b) { return (a + b - 1) / b; };

  // weights -> bf16
  cvt_f32_bf16_kernel<<<cdiv(D * D, 256), 256, 0, stream>>>(pq, wqB, D * D);
  cvt_f32_bf16_kernel<<<cdiv(D * D, 256), 256, 0, stream>>>(pk, wkB, D * D);
  cvt_f32_bf16_kernel<<<cdiv(D * D, 256), 256, 0, stream>>>(pv, wvB, D * D);
  cvt_f32_bf16_kernel<<<cdiv(D * D, 256), 256, 0, stream>>>(po, woB, D * D);
  cvt_f32_bf16_kernel<<<cdiv(DF * D, 256), 256, 0, stream>>>(w1, w1B, DF * D);
  cvt_f32_bf16_kernel<<<cdiv(DF * D, 256), 256, 0, stream>>>(w3, w3B, DF * D);
  cvt_f32_bf16_kernel<<<cdiv(D * DF, 256), 256, 0, stream>>>(w2, w2B, D * DF);

  // attention sublayer
  rmsnorm_kernel<<<M, 256, 0, stream>>>(x, ln1, h1, D);
  gemm_bf16_kernel<false><<<dim3(D / 128, M / 128), 256, 0, stream>>>(
      h1, wqB, qf, nullptr, M, D, D);
  gemm_bf16_kernel<false><<<dim3(D / 128, M / 128), 256, 0, stream>>>(
      h1, wkB, kf, nullptr, M, D, D);
  gemm_bf16_kernel<false><<<dim3(D / 128, M / 128), 256, 0, stream>>>(
      h1, wvB, vf, nullptr, M, D, D);

  int nrope = M * H * 32;  // pairs
  rope_kernel<<<cdiv(nrope, 256), 256, 0, stream>>>(qf, qb, nrope, 0.125f);
  rope_kernel<<<cdiv(nrope, 256), 256, 0, stream>>>(kf, kb, nrope, 1.0f);
  vtrans_kernel<<<cdiv(M * D, 256), 256, 0, stream>>>(vf, vT, M * D);

  flash_attn_kernel<<<dim3(S / 128, B * H), 256, 0, stream>>>(qb, kb, vT, ctx);

  gemm_bf16_kernel<true><<<dim3(D / 128, M / 128), 256, 0, stream>>>(
      ctx, woB, xmid, x, M, D, D);  // + residual x

  // FFN sublayer
  rmsnorm_kernel<<<M, 256, 0, stream>>>(xmid, ln2, h2, D);
  gemm_bf16_kernel<false><<<dim3(DF / 128, M / 128), 256, 0, stream>>>(
      h2, w1B, u1, nullptr, M, DF, D);
  gemm_bf16_kernel<false><<<dim3(DF / 128, M / 128), 256, 0, stream>>>(
      h2, w3B, u3, nullptr, M, DF, D);
  silu_mul_kernel<<<cdiv(M * DF, 256), 256, 0, stream>>>(u1, u3, gb, M * DF);
  gemm_bf16_kernel<true><<<dim3(D / 128, M / 128), 256, 0, stream>>>(
      gb, w2B, out, xmid, M, D, DF);  // + residual xmid -> d_out
}
